// GaussianGradientLSTMPolicy_27324581937666
// MI455X (gfx1250) — matrix operand order repaired and checked by reading, not compiled, not run on hardware
//
#include <hip/hip_runtime.h>
#include <math.h>

// ---------------------------------------------------------------------------
// GaussianGradientLSTMPolicy forward for MI455X (gfx1250, wave32, WMMA).
//
//   k_prep   : W1/[Wm;Ws] -> bf16; Wc = [W_ih | W_hh] bf16; bias_c = b_ih + b_hh
//   k_cvt_x  : x (f32) -> bf16 once
//   k_zero   : zero h_bf buffer0 / cbuf / barrier counter / hidden_outputs[0]
//   k_shared : shared = leaky_relu(x @ W1^T + b1)          (bf16 WMMA, f32 acc)
//   k_scan   : PERSISTENT kernel, 512 LSTM steps, grid barrier per step.
//              Block stages its 256KB slice of Wc into LDS once; inner loop is
//              ds_load_b128 B-fragments + global b128 A-fragments + WMMA.
//              Epilogue applies keep_{t+1}, records hidden_outputs[t+1],
//              stores next h as bf16 (no cvt/masking on the critical path).
//   k_heads  : [means|stds] = act(pre @ [Wm;Ws]^T + b)     (bf16 WMMA, f32 acc)
// ---------------------------------------------------------------------------

#define T_DIM 512
#define B_DIM 256
#define F_IND 128
#define H_DIM 256
#define O_DIM 32
#define GDIM  (4 * H_DIM)   // 1024
#define KCAT  (2 * H_DIM)   // 512
#define SCAN_BLOCKS 64
#define LDSW  (KCAT + 8)    // padded LDS row stride: 1040B -> bank stride 4
#define LDS_BYTES (256 * LDSW * 2)   // 266,240 B < 320KB WGP LDS

typedef __attribute__((ext_vector_type(16))) __bf16 v16bf;
typedef __attribute__((ext_vector_type(8)))  float  v8f;

union FragU {
  v16bf v;
  uint4 q[2];
  unsigned short s[16];
};

static __device__ __forceinline__ unsigned short f2bf(float f) {
  return __builtin_bit_cast(unsigned short, (__bf16)f);   // v_cvt_pk_bf16_f32
}

static __device__ __forceinline__ float sigmoidf_(float x) {
  return 1.0f / (1.0f + __expf(-x));
}

// A fragment (16x32 bf16, row-major). ISA 7.12.2:
// lanes 0-15: M=lane, VGPR0-3=K[0..7], VGPR4-7=K[16..23]; lanes16-31: K+=8/+24.
static __device__ __forceinline__ v16bf load_a_bf16(const unsigned short* __restrict__ A,
                                                    int ld, int m, int kbase) {
  FragU u;
  const unsigned short* p = A + (size_t)m * ld + kbase;
  u.q[0] = *(const uint4*)(p);
  u.q[1] = *(const uint4*)(p + 16);
  return u.v;
}

// B fragment (32x16 bf16) from row-major W[N,K]: lane n reads 32B of row n.
static __device__ __forceinline__ v16bf load_b_bf16(const unsigned short* __restrict__ W,
                                                    int ldw, int n0, int k0, int lane) {
  FragU u;
  const int n = n0 + (lane & 15);
  const int kk = k0 + ((lane >> 4) << 3);
  const unsigned short* p = W + (size_t)n * ldw + kk;
  u.q[0] = *(const uint4*)(p);
  u.q[1] = *(const uint4*)(p + 16);
  return u.v;
}

// B fragment from LDS (padded row stride LDSW) -> ds_load_b128 x2.
static __device__ __forceinline__ v16bf load_b_lds(const unsigned short* Wl,
                                                   int row0, int k0, int lane) {
  FragU u;
  const int r = row0 + (lane & 15);
  const int kk = k0 + ((lane >> 4) << 3);
  const unsigned short* p = Wl + r * LDSW + kk;
  u.q[0] = *(const uint4*)(p);
  u.q[1] = *(const uint4*)(p + 16);
  return u.v;
}

#define WMMA_BF16(a, b, c) \
  __builtin_amdgcn_wmma_f32_16x16x32_bf16(false, (a), false, (b), (short)0, (c), false, false)

// Monotonic grid barrier (counter zeroed once per kernel_launch call).
static __device__ __forceinline__ void grid_barrier(unsigned* __restrict__ cnt,
                                                    unsigned target) {
  __syncthreads();
  if (threadIdx.x == 0) {
    __hip_atomic_fetch_add(cnt, 1u, __ATOMIC_RELEASE, __HIP_MEMORY_SCOPE_AGENT);
    while (__hip_atomic_load(cnt, __ATOMIC_ACQUIRE, __HIP_MEMORY_SCOPE_AGENT) < target) {
      __builtin_amdgcn_s_sleep(2);
    }
  }
  __syncthreads();
}

// ---------------------------------------------------------------------------
__global__ void k_prep(const float* __restrict__ W1, const float* __restrict__ W_ih,
                       const float* __restrict__ W_hh, const float* __restrict__ b_ih,
                       const float* __restrict__ b_hh, const float* __restrict__ Wm,
                       const float* __restrict__ Ws,
                       unsigned short* __restrict__ W1b, unsigned short* __restrict__ Wc,
                       unsigned short* __restrict__ Wms, float* __restrict__ bias_c) {
  const int tid = blockIdx.x * blockDim.x + threadIdx.x;
  if (tid < GDIM * KCAT) {
    const int row = tid >> 9, col = tid & 511;
    const float v = (col < H_DIM) ? W_ih[row * H_DIM + col]
                                  : W_hh[row * H_DIM + (col - H_DIM)];
    Wc[tid] = f2bf(v);
  }
  if (tid < H_DIM * F_IND) W1b[tid] = f2bf(W1[tid]);
  if (tid < 64 * H_DIM) {
    const int row = tid >> 8, col = tid & 255;
    const float v = (row < O_DIM) ? Wm[row * H_DIM + col]
                                  : Ws[(row - O_DIM) * H_DIM + col];
    Wms[tid] = f2bf(v);
  }
  if (tid < GDIM) bias_c[tid] = b_ih[tid] + b_hh[tid];
}

__global__ void k_cvt_x(const float* __restrict__ x, unsigned short* __restrict__ xb,
                        int n4) {
  const int i = blockIdx.x * blockDim.x + threadIdx.x;
  if (i < n4) {
    const float4 v = ((const float4*)x)[i];
    ushort4 o;
    o.x = f2bf(v.x); o.y = f2bf(v.y); o.z = f2bf(v.z); o.w = f2bf(v.w);
    ((ushort4*)xb)[i] = o;
  }
}

__global__ void k_zero(float* __restrict__ p, int n) {
  const int i = blockIdx.x * blockDim.x + threadIdx.x;
  if (i < n) p[i] = 0.0f;
}

// shared = leaky_relu(x @ W1^T + b1) : M=131072, N=256, K=128
__global__ __launch_bounds__(128) void k_shared(const unsigned short* __restrict__ xb,
                                                const unsigned short* __restrict__ W1b,
                                                const float* __restrict__ b1,
                                                unsigned short* __restrict__ shared_out) {
  const int lane = threadIdx.x & 31;
  const int wave = blockIdx.x * 4 + (threadIdx.x >> 5);
  const int mt = wave >> 4, nt = wave & 15;
  const int row0 = mt * 16, n0 = nt * 16;
  const int mA = row0 + (lane & 15);
  const int kA = (lane >> 4) << 3;
  v8f acc = {};
#pragma unroll
  for (int k = 0; k < F_IND; k += 32) {
    const v16bf a = load_a_bf16(xb, F_IND, mA, k + kA);
    const v16bf b = load_b_bf16(W1b, F_IND, n0, k, lane);
    acc = WMMA_BF16(a, b, acc);
  }
  const int col = n0 + (lane & 15);
  const float bias = b1[col];
  const int rbase = row0 + ((lane >> 4) << 3);
#pragma unroll
  for (int r = 0; r < 8; ++r) {
    float v = acc[r] + bias;
    v = (v > 0.0f) ? v : 0.01f * v;
    shared_out[(size_t)(rbase + r) * H_DIM + col] = f2bf(v);
  }
}

// Persistent LSTM scan: 64 blocks x 4 waves. Block b -> tm = b>>2 (batch tile),
// column group tng = b&3; wave w -> tn = tng*4+w. Block stages Wc rows
// {g*256 + tng*64 .. +63, g=0..3} (256KB) into LDS once; 512 steps reuse them.
__global__ __launch_bounds__(128) void k_scan(const unsigned short* __restrict__ shared_bf,
                                              const unsigned short* __restrict__ Wc,
                                              const float* __restrict__ bias_c,
                                              const int* __restrict__ dones,
                                              unsigned short* __restrict__ hbf,  // [2][B*H]
                                              float* __restrict__ cbuf,
                                              unsigned short* __restrict__ pre_bf,
                                              float* __restrict__ hidden_out,
                                              unsigned* __restrict__ barrier_cnt) {
  extern __shared__ unsigned short ldsW[];          // [256][LDSW]
  const int lane = threadIdx.x & 31;
  const int widx = threadIdx.x >> 5;
  const int tm = blockIdx.x >> 2;                   // 0..15 batch tile
  const int tng = blockIdx.x & 3;                   // column group
  const int tn = tng * 4 + widx;                    // 0..15 hidden tile
  const int row0 = tm * 16, n0 = tn * 16;
  const int mA = row0 + (lane & 15);
  const int kA = (lane >> 4) << 3;
  const int col = n0 + (lane & 15);
  const float bi  = bias_c[col];
  const float bff = bias_c[col + H_DIM];
  const float bg  = bias_c[col + 2 * H_DIM];
  const float bo  = bias_c[col + 3 * H_DIM];
  const int rbase = row0 + ((lane >> 4) << 3);
  const int nloc = widx * 16;                       // wave's base row inside group

  // ---- stage this block's Wc slice into LDS (once) ----
  for (int i = threadIdx.x; i < 4 * 64 * 64; i += blockDim.x) {
    const int lrow = i >> 6;                        // 0..255
    const int seg = (i & 63) << 3;                  // 8-elem segment
    const int g = lrow >> 6;                        // gate
    const int rl = lrow & 63;
    const unsigned short* src = Wc + ((size_t)(g * H_DIM + tng * 64 + rl)) * KCAT + seg;
    *(uint4*)&ldsW[lrow * LDSW + seg] = *(const uint4*)src;
  }
  __syncthreads();

  for (int t = 0; t < T_DIM; ++t) {
    const unsigned short* sh_t = shared_bf + (size_t)t * B_DIM * H_DIM;
    const unsigned short* h_in = hbf + (size_t)(t & 1) * B_DIM * H_DIM;

    v8f acc0 = {}, acc1 = {}, acc2 = {}, acc3 = {};
#pragma unroll
    for (int k = 0; k < KCAT; k += 32) {
      const v16bf a = (k < H_DIM) ? load_a_bf16(sh_t, H_DIM, mA, k + kA)
                                  : load_a_bf16(h_in, H_DIM, mA, (k - H_DIM) + kA);
      acc0 = WMMA_BF16(a, load_b_lds(ldsW,           nloc, k, lane), acc0);
      acc1 = WMMA_BF16(a, load_b_lds(ldsW,  64 + nloc, k, lane), acc1);
      acc2 = WMMA_BF16(a, load_b_lds(ldsW, 128 + nloc, k, lane), acc2);
      acc3 = WMMA_BF16(a, load_b_lds(ldsW, 192 + nloc, k, lane), acc3);
    }

    const int t1 = t + 1;
    unsigned short* h_next = hbf + (size_t)(t1 & 1) * B_DIM * H_DIM;
    float* hid_h = hidden_out + (size_t)t1 * 2 * B_DIM * H_DIM;
    float* hid_c = hid_h + (size_t)B_DIM * H_DIM;
    const int* dn = dones + (size_t)t1 * B_DIM;

#pragma unroll
    for (int r = 0; r < 8; ++r) {
      const int m = rbase + r;
      const size_t idx = (size_t)m * H_DIM + col;
      const float cm = cbuf[idx];                   // masked with keep_t already
      const float gi = sigmoidf_(acc0[r] + bi);
      const float gf = sigmoidf_(acc1[r] + bff);
      const float gg = tanhf(acc2[r] + bg);
      const float go = sigmoidf_(acc3[r] + bo);
      const float c2 = gf * cm + gi * gg;
      const float h2 = go * tanhf(c2);
      const float pre = (h2 > 0.0f) ? h2 : 0.01f * h2;
      pre_bf[((size_t)t * B_DIM + m) * H_DIM + col] = f2bf(pre);
      if (t1 < T_DIM) {                             // pre-mask next step's state
        const float keepn = dn[m] ? 0.0f : 1.0f;
        const float hm2 = h2 * keepn;
        const float cm2 = c2 * keepn;
        hid_h[idx] = hm2;                           // hidden_outputs[t+1]
        hid_c[idx] = cm2;
        h_next[idx] = f2bf(hm2);                    // bf16 A operand for t+1
        cbuf[idx] = cm2;
      }
    }

    if (t1 < T_DIM) {                               // hide next shared_t fetch
      __builtin_prefetch(shared_bf + ((size_t)t1 * B_DIM + mA) * H_DIM + kA, 0, 3);
      __builtin_prefetch(shared_bf + ((size_t)t1 * B_DIM + mA) * H_DIM + kA + 128, 0, 3);
    }
    grid_barrier(barrier_cnt, (unsigned)(SCAN_BLOCKS * t1));
  }
}

// heads: [means|stds] = act(pre @ [Wm;Ws]^T + [bm;bs]) : M=131072, N=64, K=256
__global__ __launch_bounds__(128) void k_heads(const unsigned short* __restrict__ pre_bf,
                                               const unsigned short* __restrict__ Wms,
                                               const float* __restrict__ bm,
                                               const float* __restrict__ bs,
                                               float* __restrict__ means,
                                               float* __restrict__ stds) {
  const int lane = threadIdx.x & 31;
  const int wave = blockIdx.x * 4 + (threadIdx.x >> 5);
  const int mt = wave >> 2, nt = wave & 3;
  const int row0 = mt * 16, n0 = nt * 16;
  const int mA = row0 + (lane & 15);
  const int kA = (lane >> 4) << 3;
  v8f acc = {};
#pragma unroll
  for (int k = 0; k < H_DIM; k += 32) {
    const v16bf a = load_a_bf16(pre_bf, H_DIM, mA, k + kA);
    const v16bf b = load_b_bf16(Wms, H_DIM, n0, k, lane);
    acc = WMMA_BF16(a, b, acc);
  }
  const int col = n0 + (lane & 15);                 // 0..63
  const int rbase = row0 + ((lane >> 4) << 3);
#pragma unroll
  for (int r = 0; r < 8; ++r) {
    const int row = rbase + r;
    const float v = acc[r];
    if (col < O_DIM) {
      means[(size_t)row * O_DIM + col] = tanhf(v + bm[col]);
    } else {
      const float xs = v + bs[col - O_DIM];
      const float sp = (xs > 20.0f) ? xs : log1pf(__expf(xs));
      stds[(size_t)row * O_DIM + (col - O_DIM)] = sp + 1e-6f;
    }
  }
}

// ---------------------------------------------------------------------------
extern "C" void kernel_launch(void* const* d_in, const int* in_sizes, int n_in,
                              void* d_out, int out_size, void* d_ws, size_t ws_size,
                              hipStream_t stream) {
  (void)in_sizes; (void)n_in; (void)out_size; (void)ws_size;
  const float* x    = (const float*)d_in[0];
  const int*   dones= (const int*)d_in[1];
  const float* W1   = (const float*)d_in[2];
  const float* b1   = (const float*)d_in[3];
  const float* W_ih = (const float*)d_in[4];
  const float* W_hh = (const float*)d_in[5];
  const float* b_ih = (const float*)d_in[6];
  const float* b_hh = (const float*)d_in[7];
  const float* Wm   = (const float*)d_in[8];
  const float* bm   = (const float*)d_in[9];
  const float* Ws   = (const float*)d_in[10];
  const float* bs   = (const float*)d_in[11];

  char* ws = (char*)d_ws;
  size_t off = 0;
  auto alloc = [&](size_t bytes) -> char* {
    char* p = ws + off;
    off += (bytes + 255) & ~(size_t)255;
    return p;
  };
  unsigned short* x_bf      = (unsigned short*)alloc((size_t)T_DIM * B_DIM * F_IND * 2);
  unsigned short* shared_bf = (unsigned short*)alloc((size_t)T_DIM * B_DIM * H_DIM * 2);
  unsigned short* pre_bf    = (unsigned short*)alloc((size_t)T_DIM * B_DIM * H_DIM * 2);
  unsigned short* Wc        = (unsigned short*)alloc((size_t)GDIM * KCAT * 2);
  unsigned short* W1b       = (unsigned short*)alloc((size_t)H_DIM * F_IND * 2);
  unsigned short* Wms       = (unsigned short*)alloc((size_t)64 * H_DIM * 2);
  float*          bias_c    = (float*)alloc((size_t)GDIM * 4);
  unsigned short* hbf       = (unsigned short*)alloc((size_t)2 * B_DIM * H_DIM * 2);
  float*          cbuf      = (float*)alloc((size_t)B_DIM * H_DIM * 4);
  unsigned*       bar_cnt   = (unsigned*)alloc(256);

  float* means  = (float*)d_out;
  float* stds   = means + (size_t)T_DIM * B_DIM * O_DIM;
  float* hidden = stds + (size_t)T_DIM * B_DIM * O_DIM;

  // weight prep (max range: 1024*512 = 524288 threads)
  k_prep<<<2048, 256, 0, stream>>>(W1, W_ih, W_hh, b_ih, b_hh, Wm, Ws,
                                   W1b, Wc, Wms, bias_c);
  // x -> bf16 (4 elems/thread)
  k_cvt_x<<<(T_DIM * B_DIM * F_IND / 4 + 255) / 256, 256, 0, stream>>>(
      x, x_bf, T_DIM * B_DIM * F_IND / 4);
  // zero: h_bf buffer0 (as 32768 floats), cbuf, barrier counter, hidden[0]
  k_zero<<<128, 256, 0, stream>>>((float*)hbf, B_DIM * H_DIM / 2);
  k_zero<<<256, 256, 0, stream>>>(cbuf, B_DIM * H_DIM);
  k_zero<<<1, 32, 0, stream>>>((float*)bar_cnt, 1);
  k_zero<<<512, 256, 0, stream>>>(hidden, 2 * B_DIM * H_DIM);
  // front GEMM: 131072 x 256 x 128
  k_shared<<<32768, 128, 0, stream>>>(x_bf, W1b, b1, shared_bf);
  // persistent serial scan (512 steps, grid barrier per step, 256KB LDS/WGP)
  k_scan<<<SCAN_BLOCKS, 128, LDS_BYTES, stream>>>(shared_bf, Wc, bias_c, dones,
                                                  hbf, cbuf, pre_bf, hidden, bar_cnt);
  // output heads: 131072 x 64 x 256
  k_heads<<<8192, 128, 0, stream>>>(pre_bf, Wms, bm, bs, means, stds);
}
